// RuntimePredictionGNN_62070867362011
// MI455X (gfx1250) — matrix operand order repaired and checked by reading, not compiled, not run on hardware
//
#include <hip/hip_runtime.h>
#include <hip/hip_bf16.h>

#define HD   64
#define NFD  16
#define EFD  8
#define KMSG 72
#define KUPD 128
#define NL   4

typedef _Float16 f16;
typedef __attribute__((ext_vector_type(16))) _Float16 v16h;
typedef __attribute__((ext_vector_type(8)))  _Float16 v8h;
typedef __attribute__((ext_vector_type(8)))  float    v8f;

// ---- CDNA5 WMMA 16x16x32 f16 fragment index maps (ISA 7.12.2) ----
__device__ __forceinline__ void bmap(int lane, int i, int& k, int& n) {
  n = lane & 15;
  k = ((lane < 16) ? 0 : 8) + i + ((i < 8) ? 0 : 8);
}
__device__ __forceinline__ void cmap(int lane, int r, int& m, int& n) {
  n = lane & 15;
  m = r + ((lane < 16) ? 0 : 8);
}

// Build a 16x32 A fragment slice from a row-major f16 row: halves {b..b+7, b+16..b+23}
// with b = kc*32 + (lane<16 ? 0 : 8). Row base must be 16B aligned.
__device__ __forceinline__ v16h load_a_f16row(const f16* rowp, int kc, int gsel) {
  const v8h* hp = (const v8h*)rowp;
  v8h lo = hp[kc * 4 + gsel];
  v8h hi = hp[kc * 4 + gsel + 2];
  return __builtin_shufflevector(lo, hi, 0, 1, 2, 3, 4, 5, 6, 7,
                                 8, 9, 10, 11, 12, 13, 14, 15);
}

// ------------------------------ utility kernels ------------------------------
__global__ void fill_f32(float* p, float v, int n) {
  int i = blockIdx.x * blockDim.x + threadIdx.x;
  for (; i < n; i += gridDim.x * blockDim.x) p[i] = v;
}
__global__ void fill_u32(unsigned* p, unsigned v, int n) {
  int i = blockIdx.x * blockDim.x + threadIdx.x;
  for (; i < n; i += gridDim.x * blockDim.x) p[i] = v;
}
__global__ void cvt_f32_to_f16(const float* __restrict__ s, f16* __restrict__ d, int n) {
  int i = blockIdx.x * blockDim.x + threadIdx.x;
  for (; i < n; i += gridDim.x * blockDim.x) d[i] = (f16)s[i];
}

__global__ void deg_kernel(const int* __restrict__ dst, float* __restrict__ degf, int E) {
  int i = blockIdx.x * blockDim.x + threadIdx.x;
  for (; i < E; i += gridDim.x * blockDim.x) atomicAdd(&degf[dst[i]], 1.0f);
}

// ------------------------------ node embedding ------------------------------
__global__ void embed_kernel(const float* __restrict__ x, const float* __restrict__ W0,
                             const float* __restrict__ b0, const float* __restrict__ g,
                             const float* __restrict__ be, float* __restrict__ h,
                             f16* __restrict__ hHalf, int N) {
  int wave = (blockIdx.x * blockDim.x + threadIdx.x) >> 5;
  int lane = threadIdx.x & 31;
  if (wave >= N) return;
  float v0 = b0[lane], v1 = b0[lane + 32];
  #pragma unroll
  for (int i = 0; i < NFD; ++i) {
    float xv = x[wave * NFD + i];
    v0 += xv * W0[i * HD + lane];
    v1 += xv * W0[i * HD + lane + 32];
  }
  v0 = fmaxf(v0, 0.f); v1 = fmaxf(v1, 0.f);
  float s = v0 + v1, ss = v0 * v0 + v1 * v1;
  #pragma unroll
  for (int off = 16; off > 0; off >>= 1) {
    s  += __shfl_xor(s,  off, 32);
    ss += __shfl_xor(ss, off, 32);
  }
  float mean = s * (1.f / HD);
  float var  = ss * (1.f / HD) - mean * mean;
  float inv  = rsqrtf(var + 1e-5f);
  float o0 = (v0 - mean) * inv * g[lane]      + be[lane];
  float o1 = (v1 - mean) * inv * g[lane + 32] + be[lane + 32];
  h[wave * HD + lane]      = o0;
  h[wave * HD + lane + 32] = o1;
  hHalf[wave * HD + lane]      = (f16)o0;
  hHalf[wave * HD + lane + 32] = (f16)o1;
}

// ------------------------------ edge message GEMM (WMMA) ------------------------------
__global__ __launch_bounds__(256) void
msg_kernel(const f16* __restrict__ hHalf, const f16* __restrict__ WmsgH,
           const float* __restrict__ bmsg, const float* __restrict__ gateEmb,
           const int* __restrict__ srcIdx, const int* __restrict__ dstIdx,
           const float* __restrict__ edgeAttr, const int* __restrict__ gateType,
           float* __restrict__ agg, int E) {
  const int lane   = threadIdx.x & 31;
  const int wave   = (blockIdx.x * blockDim.x + threadIdx.x) >> 5;
  const int nwaves = (gridDim.x * blockDim.x) >> 5;
  const int ntiles = (E + 15) >> 4;
  const int gsel   = (lane < 16) ? 0 : 1;
  // wave-uniform copies (lets the compiler keep tile control flow scalar)
  const int waveU   = __builtin_amdgcn_readfirstlane(wave);
  const int nwavesU = __builtin_amdgcn_readfirstlane(nwaves);

  // Preload B = Wmsg (72x64 f16, K padded to 96): 3 K-chunks x 4 N-tiles, in VGPRs.
  v16h bfrag[3][4];
  #pragma unroll
  for (int kc = 0; kc < 3; ++kc)
    #pragma unroll
    for (int nt = 0; nt < 4; ++nt)
      #pragma unroll
      for (int i = 0; i < 16; ++i) {
        int k, n; bmap(lane, i, k, n);
        int kk = kc * 32 + k;
        bfrag[kc][nt][i] = (kk < KMSG) ? WmsgH[kk * HD + nt * 16 + n] : (f16)0.f;
      }
  float bm[4];
  #pragma unroll
  for (int nt = 0; nt < 4; ++nt) bm[nt] = bmsg[nt * 16 + (lane & 15)];

  for (int tile = waveU; tile < ntiles; tile += nwavesU) {
    const int ebase = tile << 4;   // scalar
    int rowm  = ebase + (lane & 15);
    int myrow = rowm < E ? rowm : E - 1;
    int s     = srcIdx[myrow];
    int gt_my = gateType[myrow];
    int d_my  = dstIdx[myrow];

    v8f c[4] = {};
    // K-chunks 0,1: gathered node features (row-major f16, vector loads)
    #pragma unroll
    for (int kc = 0; kc < 2; ++kc) {
      v16h a = load_a_f16row(hHalf + (size_t)s * HD, kc, gsel);
      #pragma unroll
      for (int nt = 0; nt < 4; ++nt)
        c[nt] = __builtin_amdgcn_wmma_f32_16x16x32_f16(
            false, a, false, bfrag[kc][nt], (short)0, c[nt], false, false);
    }
    // K-chunk 2: edge_attr (8 f32 as 2x float4; lanes>=16 pad 0)
    {
      const float4* eap = (const float4*)(edgeAttr + (size_t)myrow * EFD);
      v16h a;
      #pragma unroll
      for (int i = 0; i < 16; ++i) a[i] = (f16)0.f;
      if (lane < 16) {
        float4 e0 = eap[0], e1 = eap[1];
        a[0] = (f16)e0.x; a[1] = (f16)e0.y; a[2] = (f16)e0.z; a[3] = (f16)e0.w;
        a[4] = (f16)e1.x; a[5] = (f16)e1.y; a[6] = (f16)e1.z; a[7] = (f16)e1.w;
      }
      #pragma unroll
      for (int nt = 0; nt < 4; ++nt)
        c[nt] = __builtin_amdgcn_wmma_f32_16x16x32_f16(
            false, a, false, bfrag[2][nt], (short)0, c[nt], false, false);
    }
    // Epilogue: bias + gate embedding + relu + scatter-add.
    if (ebase + 16 <= E) {            // scalar condition -> s_cbranch, no exec dance
      #pragma unroll
      for (int r = 0; r < 8; ++r) {
        int m, n0; cmap(lane, r, m, n0);
        int gt = __shfl(gt_my, m, 32);
        int d  = __shfl(d_my,  m, 32);
        const float* gerow = gateEmb + (size_t)gt * HD;
        float* arow = agg + (size_t)d * HD;
        #pragma unroll
        for (int nt = 0; nt < 4; ++nt) {
          int col = nt * 16 + n0;
          float v = fmaxf(c[nt][r] + bm[nt] + gerow[col], 0.f);
          atomicAdd(&arow[col], v);
        }
      }
    } else {                          // tail tile
      #pragma unroll
      for (int r = 0; r < 8; ++r) {
        int m, n0; cmap(lane, r, m, n0);
        int e  = ebase + m;
        int gt = __shfl(gt_my, m, 32);
        int d  = __shfl(d_my,  m, 32);
        if (e < E) {
          #pragma unroll
          for (int nt = 0; nt < 4; ++nt) {
            int col = nt * 16 + n0;
            float v = fmaxf(c[nt][r] + bm[nt] + gateEmb[(size_t)gt * HD + col], 0.f);
            atomicAdd(&agg[(size_t)d * HD + col], v);
          }
        }
      }
    }
  }
}

// ------------------------------ node update GEMM (WMMA) ------------------------------
__global__ __launch_bounds__(256) void
upd_kernel(float* __restrict__ h, f16* __restrict__ hHalf,
           const f16* __restrict__ WupdH, const float* __restrict__ bupd,
           float* __restrict__ agg, const float* __restrict__ degf, int N) {
  const int lane   = threadIdx.x & 31;
  const int wave   = (blockIdx.x * blockDim.x + threadIdx.x) >> 5;
  const int nwaves = (gridDim.x * blockDim.x) >> 5;
  const int ntiles = (N + 15) >> 4;
  const int gsel   = (lane < 16) ? 0 : 1;
  const int waveU   = __builtin_amdgcn_readfirstlane(wave);
  const int nwavesU = __builtin_amdgcn_readfirstlane(nwaves);

  v16h bfrag[4][4];  // K=128: 4 chunks x 4 N-tiles
  #pragma unroll
  for (int kc = 0; kc < 4; ++kc)
    #pragma unroll
    for (int nt = 0; nt < 4; ++nt)
      #pragma unroll
      for (int i = 0; i < 16; ++i) {
        int k, n; bmap(lane, i, k, n);
        bfrag[kc][nt][i] = WupdH[(kc * 32 + k) * HD + nt * 16 + n];
      }
  float bu[4];
  #pragma unroll
  for (int nt = 0; nt < 4; ++nt) bu[nt] = bupd[nt * 16 + (lane & 15)];

  for (int tile = waveU; tile < ntiles; tile += nwavesU) {
    const int nbase = tile << 4;   // scalar
    int rowm = nbase + (lane & 15);
    int node = rowm < N ? rowm : N - 1;
    float dginv = 1.f / fmaxf(degf[node], 1.f);

    v8f c[4] = {};
    #pragma unroll
    for (int kc = 0; kc < 2; ++kc) {
      v16h a = load_a_f16row(hHalf + (size_t)node * HD, kc, gsel);
      #pragma unroll
      for (int nt = 0; nt < 4; ++nt)
        c[nt] = __builtin_amdgcn_wmma_f32_16x16x32_f16(
            false, a, false, bfrag[kc][nt], (short)0, c[nt], false, false);
    }
    const float4* ap = (const float4*)(agg + (size_t)node * HD);
    #pragma unroll
    for (int kc = 2; kc < 4; ++kc) {
      int fb = (kc - 2) * 8 + gsel * 2;
      float4 q0 = ap[fb],     q1 = ap[fb + 1];
      float4 q2 = ap[fb + 4], q3 = ap[fb + 5];
      v16h a;
      a[0]  = (f16)(q0.x * dginv); a[1]  = (f16)(q0.y * dginv);
      a[2]  = (f16)(q0.z * dginv); a[3]  = (f16)(q0.w * dginv);
      a[4]  = (f16)(q1.x * dginv); a[5]  = (f16)(q1.y * dginv);
      a[6]  = (f16)(q1.z * dginv); a[7]  = (f16)(q1.w * dginv);
      a[8]  = (f16)(q2.x * dginv); a[9]  = (f16)(q2.y * dginv);
      a[10] = (f16)(q2.z * dginv); a[11] = (f16)(q2.w * dginv);
      a[12] = (f16)(q3.x * dginv); a[13] = (f16)(q3.y * dginv);
      a[14] = (f16)(q3.z * dginv); a[15] = (f16)(q3.w * dginv);
      #pragma unroll
      for (int nt = 0; nt < 4; ++nt)
        c[nt] = __builtin_amdgcn_wmma_f32_16x16x32_f16(
            false, a, false, bfrag[kc][nt], (short)0, c[nt], false, false);
    }
    // Epilogue (WMMA consumed the agg/h loads, so these stores cannot overtake them)
    if (nbase + 16 <= N) {            // scalar condition
      #pragma unroll
      for (int r = 0; r < 8; ++r) {
        int m, n0; cmap(lane, r, m, n0);
        int n2 = nbase + m;
        float* hrow  = h + (size_t)n2 * HD;
        f16*   hhrow = hHalf + (size_t)n2 * HD;
        float* arow  = agg + (size_t)n2 * HD;
        #pragma unroll
        for (int nt = 0; nt < 4; ++nt) {
          int col = nt * 16 + n0;
          float hn = fmaxf(c[nt][r] + bu[nt], 0.f);
          float nh = hrow[col] + hn;
          hrow[col]  = nh;
          hhrow[col] = (f16)nh;
          arow[col]  = 0.f;   // ready for next layer
        }
      }
    } else {                          // tail tile
      #pragma unroll
      for (int r = 0; r < 8; ++r) {
        int m, n0; cmap(lane, r, m, n0);
        int n2 = nbase + m;
        if (n2 < N) {
          #pragma unroll
          for (int nt = 0; nt < 4; ++nt) {
            int col = nt * 16 + n0;
            float hn = fmaxf(c[nt][r] + bu[nt], 0.f);
            float nh = h[(size_t)n2 * HD + col] + hn;
            h[(size_t)n2 * HD + col]     = nh;
            hHalf[(size_t)n2 * HD + col] = (f16)nh;
            agg[(size_t)n2 * HD + col]   = 0.f;
          }
        }
      }
    }
  }
}

// ------------------------------ graph pooling ------------------------------
__global__ void pool_kernel(const float* __restrict__ h, const int* __restrict__ batch,
                            float* __restrict__ psum, unsigned* __restrict__ pmaxb,
                            float* __restrict__ pcnt, int N) {
  int i = blockIdx.x * blockDim.x + threadIdx.x;
  int total = N * HD;
  for (; i < total; i += gridDim.x * blockDim.x) {
    int node = i >> 6, col = i & 63;
    int b = batch[node];
    float v = h[i];
    atomicAdd(&psum[b * HD + col], v);
    if (v >= 0.f) atomicMax((int*)&pmaxb[b * HD + col], __float_as_int(v));
    else          atomicMin(&pmaxb[b * HD + col], __float_as_uint(v));
    if (col == 0) atomicAdd(&pcnt[b], 1.f);
  }
}

// ------------------------------ per-graph head ------------------------------
__global__ __launch_bounds__(128) void
head_kernel(const float* __restrict__ psum, const unsigned* __restrict__ pmaxb,
            const float* __restrict__ pcnt, const float* __restrict__ gf, int gfdim,
            const int* __restrict__ tcl,
            const float* __restrict__ Wg, const float* __restrict__ bg,
            const float* __restrict__ lngg, const float* __restrict__ lngb,
            const float* __restrict__ embth, const float* __restrict__ Wt,
            const float* __restrict__ bt,
            const float* __restrict__ Wc1, const float* __restrict__ bc1,
            const float* __restrict__ lncg, const float* __restrict__ lncb,
            const float* __restrict__ Wc2, const float* __restrict__ bc2,
            const float* __restrict__ Wh1, const float* __restrict__ bh1,
            const float* __restrict__ Wh2, const float* __restrict__ bh2,
            float* __restrict__ out) {
  __shared__ float c[288];
  __shared__ float c1[128];
  __shared__ float c2s[64];
  __shared__ float t1[32];
  __shared__ float gtmp[64];
  __shared__ float stats[2];
  const int b = blockIdx.x, t = threadIdx.x;
  const float cnt = pcnt[b];

  if (t < 64) {
    float s  = psum[b * HD + t];
    float mx = __uint_as_float(pmaxb[b * HD + t]);
    if (cnt <= 0.f || mx == -__builtin_huge_valf()) mx = 0.f;
    c[t]       = s / fmaxf(cnt, 1.f);
    c[64 + t]  = mx;
    c[128 + t] = s;
    float v = bg[t];
    for (int i = 0; i < gfdim; ++i) v += gf[b * gfdim + i] * Wg[i * HD + t];
    gtmp[t] = fmaxf(v, 0.f);
  }
  if (t < 32) {
    int cls = tcl[b];
    float v = bt[t];
    for (int i = 0; i < 32; ++i) v += embth[cls * 32 + i] * Wt[i * 32 + t];
    c[256 + t] = fmaxf(v, 0.f);
  }
  __syncthreads();
  if (t == 0) {
    float s = 0.f, ss = 0.f;
    for (int i = 0; i < 64; ++i) { s += gtmp[i]; ss += gtmp[i] * gtmp[i]; }
    float m = s / 64.f;
    stats[0] = m; stats[1] = rsqrtf(ss / 64.f - m * m + 1e-5f);
  }
  __syncthreads();
  if (t < 64) c[192 + t] = (gtmp[t] - stats[0]) * stats[1] * lngg[t] + lngb[t];
  __syncthreads();
  {
    float v = bc1[t];
    for (int i = 0; i < 288; ++i) v += c[i] * Wc1[i * 128 + t];
    c1[t] = fmaxf(v, 0.f);
  }
  __syncthreads();
  if (t == 0) {
    float s = 0.f, ss = 0.f;
    for (int i = 0; i < 128; ++i) { s += c1[i]; ss += c1[i] * c1[i]; }
    float m = s / 128.f;
    stats[0] = m; stats[1] = rsqrtf(ss / 128.f - m * m + 1e-5f);
  }
  __syncthreads();
  float c1n = (c1[t] - stats[0]) * stats[1] * lncg[t] + lncb[t];
  __syncthreads();
  c1[t] = c1n;
  __syncthreads();
  if (t < 64) {
    float v = bc2[t];
    for (int i = 0; i < 128; ++i) v += c1[i] * Wc2[i * 64 + t];
    c2s[t] = fmaxf(v, 0.f);
  }
  __syncthreads();
  if (t < 32) {
    float v = bh1[t];
    for (int i = 0; i < 64; ++i) v += c2s[i] * Wh1[i * 32 + t];
    t1[t] = fmaxf(v, 0.f);
  }
  __syncthreads();
  if (t == 0) {
    float v = bh2[0];
    for (int i = 0; i < 32; ++i) v += t1[i] * Wh2[i];
    out[b] = v;
  }
}

// ------------------------------ launcher ------------------------------
extern "C" void kernel_launch(void* const* d_in, const int* in_sizes, int n_in,
                              void* d_out, int out_size, void* d_ws, size_t ws_size,
                              hipStream_t stream) {
  (void)n_in; (void)out_size; (void)ws_size;
  const float* x    = (const float*)d_in[0];
  const int*   ei   = (const int*)  d_in[1];
  const float* ea   = (const float*)d_in[2];
  const int*   egt  = (const int*)  d_in[3];
  const int*   batv = (const int*)  d_in[4];
  const float* gfe  = (const float*)d_in[5];
  const int*   tcl  = (const int*)  d_in[6];
  const float* W0   = (const float*)d_in[7];
  const float* b0   = (const float*)d_in[8];
  const float* ln1g = (const float*)d_in[9];
  const float* ln1b = (const float*)d_in[10];
  const float* Wmsg = (const float*)d_in[11];
  const float* bmsg = (const float*)d_in[12];
  const float* gemb = (const float*)d_in[13];
  const float* Wupd = (const float*)d_in[14];
  const float* bupd = (const float*)d_in[15];
  const float* Wg   = (const float*)d_in[16];
  const float* bg   = (const float*)d_in[17];
  const float* lngg = (const float*)d_in[18];
  const float* lngb = (const float*)d_in[19];
  const float* embth= (const float*)d_in[20];
  const float* Wt   = (const float*)d_in[21];
  const float* bt   = (const float*)d_in[22];
  const float* Wc1  = (const float*)d_in[23];
  const float* bc1  = (const float*)d_in[24];
  const float* lncg = (const float*)d_in[25];
  const float* lncb = (const float*)d_in[26];
  const float* Wc2  = (const float*)d_in[27];
  const float* bc2  = (const float*)d_in[28];
  const float* Wh1  = (const float*)d_in[29];
  const float* bh1  = (const float*)d_in[30];
  const float* Wh2  = (const float*)d_in[31];
  const float* bh2  = (const float*)d_in[32];

  const int N  = in_sizes[0] / NFD;
  const int E  = in_sizes[3];
  const int B  = in_sizes[6];
  const int GFdim  = in_sizes[5] / B;
  const int Gtypes = in_sizes[13] / (NL * HD);

  char* w = (char*)d_ws;
  size_t off = 0;
  auto take = [&](size_t bytes) -> char* {
    char* p = w + off; off = (off + bytes + 255) & ~(size_t)255; return p;
  };
  float*    h     = (float*)   take((size_t)N * HD * 4);
  f16*      hHalf = (f16*)     take((size_t)N * HD * 2);
  float*    agg   = (float*)   take((size_t)N * HD * 4);
  float*    degf  = (float*)   take((size_t)N * 4);
  float*    psum  = (float*)   take((size_t)B * HD * 4);
  unsigned* pmaxb = (unsigned*)take((size_t)B * HD * 4);
  float*    pcnt  = (float*)   take((size_t)B * 4);
  f16*      WmsgH = (f16*)     take((size_t)NL * KMSG * HD * 2);
  f16*      WupdH = (f16*)     take((size_t)NL * KUPD * HD * 2);

  cvt_f32_to_f16<<<64, 256, 0, stream>>>(Wmsg, WmsgH, NL * KMSG * HD);
  cvt_f32_to_f16<<<128, 256, 0, stream>>>(Wupd, WupdH, NL * KUPD * HD);
  fill_f32<<<1024, 256, 0, stream>>>(agg, 0.f, N * HD);
  fill_f32<<<64, 256, 0, stream>>>(degf, 0.f, N);
  fill_f32<<<32, 256, 0, stream>>>(psum, 0.f, B * HD);
  fill_u32<<<32, 256, 0, stream>>>(pmaxb, 0xFF800000u, B * HD);  // -inf bits
  fill_f32<<<4, 256, 0, stream>>>(pcnt, 0.f, B);

  embed_kernel<<<(N + 7) / 8, 256, 0, stream>>>(x, W0, b0, ln1g, ln1b, h, hHalf, N);
  deg_kernel<<<1024, 256, 0, stream>>>(ei + E, degf, E);

  for (int l = 0; l < NL; ++l) {
    msg_kernel<<<1280, 256, 0, stream>>>(
        hHalf, WmsgH + (size_t)l * KMSG * HD, bmsg + l * HD,
        gemb + (size_t)l * Gtypes * HD, ei, ei + E, ea, egt, agg, E);
    upd_kernel<<<512, 256, 0, stream>>>(
        h, hHalf, WupdH + (size_t)l * KUPD * HD, bupd + l * HD, agg, degf, N);
  }

  pool_kernel<<<2048, 256, 0, stream>>>(h, batv, psum, pmaxb, pcnt, N);
  head_kernel<<<B, 128, 0, stream>>>(psum, pmaxb, pcnt, gfe, GFdim, tcl,
                                     Wg, bg, lngg, lngb, embth, Wt, bt,
                                     Wc1, bc1, lncg, lncb, Wc2, bc2,
                                     Wh1, bh1, Wh2, bh2, (float*)d_out);
}
